// vig_tinytiny_54348516163872
// MI455X (gfx1250) — compile-verified, matrix operand order checked by reading, not run on hardware
//
#include <hip/hip_runtime.h>
#include <hip/hip_bf16.h>
#include <math.h>

// ---------------------------------------------------------------------------
// CDNA5 (gfx1250) ViG-tiny forward. wave32, WMMA bf16 16x16x32, f32 accum.
// 1x1-conv GEMMs stage A/B LDS tiles with the Tensor Data Mover using a
// DOUBLE-BUFFERED pipeline (issue tile t+1, s_wait_tensorcnt<=2, compute
// tile t) so the DMA overlaps the WMMA work. 3x3 convs use a division-free
// implicit-im2col gather. Weights are kept transposed (N x K) so both A and
// B WMMA fragments are contiguous-K in LDS -> ds_load_b128 pairs.
// ---------------------------------------------------------------------------

typedef __bf16 bf16_t;
typedef __attribute__((ext_vector_type(16))) __bf16 v16bf;
typedef __attribute__((ext_vector_type(8)))  float  v8f;
typedef __attribute__((ext_vector_type(4)))  unsigned int u32x4;
typedef __attribute__((ext_vector_type(8)))  int i32x8;
typedef __attribute__((ext_vector_type(4)))  int i32x4;

#define BN_INV 0.99999500003749973f  // 1/sqrt(1+1e-5)
#define TBM 64
#define TBN 64
#define TBK 32
#define TOPK 9

__device__ __forceinline__ float gelu_exact(float x) {
    return 0.5f * x * (1.0f + erff(x * 0.70710678118654752f));
}

// ---------------------------------------------------------------------------
// TDM: 2-D tile (tile_d1 rows x tile_d0 elems, bf16) global -> LDS.
// D# per CDNA5 ISA 8.3/8.4: group0 = {count=1|type=2, lds_addr, global_addr},
// group1 = {data_size=2B, tensor dims, tile dims, row stride}. Groups 2/3
// zero (2-D tensor). Issued once per wave; completion via TENSORcnt.
// ---------------------------------------------------------------------------
__device__ __forceinline__ void tdm_load_2d(unsigned lds_off, const void* gsrc,
                                            int tensor_d0, int tensor_d1,
                                            int tile_d0, int tile_d1,
                                            long long row_stride_elems)
{
    unsigned long long ga = (unsigned long long)(size_t)gsrc;
    u32x4 g0;
    g0[0] = 1u;                                   // count=1, user mode
    g0[1] = lds_off;                              // lds_addr (bytes)
    g0[2] = (unsigned)(ga & 0xffffffffu);         // global_addr[31:0]
    g0[3] = (unsigned)((ga >> 32) & 0x01ffffffu)  // global_addr[56:32]
            | 0x80000000u;                        // type=2 ("image")
    i32x8 g1;
    g1[0] = 0x00010000;                           // data_size=1 (2 bytes)
    g1[1] = (tensor_d0 & 0xffff) << 16;           // tensor_dim0[15:0]
    g1[2] = ((tensor_d0 >> 16) & 0xffff) | ((tensor_d1 & 0xffff) << 16);
    g1[3] = ((tensor_d1 >> 16) & 0xffff) | ((tile_d0 & 0xffff) << 16);
    g1[4] = (tile_d1 & 0xffff);                   // tile_dim1 (tile_dim2=0)
    g1[5] = (int)(row_stride_elems & 0xffffffffll);
    g1[6] = (int)((row_stride_elems >> 32) & 0xffffll);
    g1[7] = 0;
    i32x4 z4 = {0, 0, 0, 0};
    i32x8 z8 = {0, 0, 0, 0, 0, 0, 0, 0};
    __builtin_amdgcn_tensor_load_to_lds(g0, g1, z4, z4, z8, 0);
}

// ---------------------------------------------------------------------------
// Stem conv1: 3->384, 3x3 s2 p1, bias+BN+ReLU. K=27 (not WMMA-shaped).
// x NCHW f32 -> h1 NHWC bf16 (feeds the s2 implicit GEMM directly).
// ---------------------------------------------------------------------------
__global__ __launch_bounds__(256) void k_stem1(
    const float* __restrict__ x, const float* __restrict__ w,
    const float* __restrict__ bias, const float* __restrict__ g,
    const float* __restrict__ bb, bf16_t* __restrict__ out,
    int Bn, int Cin, int Hin, int Win, int Cout, int OH, int OW)
{
    long long gid = (long long)blockIdx.x * blockDim.x + threadIdx.x;
    long long total = (long long)Bn * OH * OW * Cout;
    if (gid >= total) return;
    int oc  = (int)(gid % Cout);
    long long pix = gid / Cout;
    int ox = (int)(pix % OW);  pix /= OW;
    int oy = (int)(pix % OH);
    int b  = (int)(pix / OH);
    float acc = bias[oc];
    for (int c = 0; c < Cin; ++c)
        for (int ky = 0; ky < 3; ++ky) {
            int iy = oy * 2 - 1 + ky;
            if ((unsigned)iy >= (unsigned)Hin) continue;
            for (int kx = 0; kx < 3; ++kx) {
                int ix = ox * 2 - 1 + kx;
                if ((unsigned)ix >= (unsigned)Win) continue;
                acc += x[(((long long)b * Cin + c) * Hin + iy) * Win + ix] *
                       w[((oc * Cin + c) * 3 + ky) * 3 + kx];
            }
        }
    float y = acc * (g[oc] * BN_INV) + bb[oc];
    out[gid] = (bf16_t)fmaxf(y, 0.0f);
}

// ---------------------------------------------------------------------------
// Weight prep: OIHW f32 -> TRANSPOSED bf16 BwT (N x K), k=((ky*kd+kx)*Cin+c)
// ---------------------------------------------------------------------------
__global__ __launch_bounds__(256) void k_prep_w(
    const float* __restrict__ w, bf16_t* __restrict__ BwT,
    int O, int Cin, int kd)
{
    long long gid = (long long)blockIdx.x * blockDim.x + threadIdx.x;
    long long total = (long long)O * Cin * kd * kd;
    if (gid >= total) return;
    int kx = (int)(gid % kd); long long t = gid / kd;
    int ky = (int)(t % kd);   t /= kd;
    int c  = (int)(t % Cin);
    int o  = (int)(t / Cin);
    int K  = Cin * kd * kd;
    BwT[(long long)o * K + (ky * kd + kx) * Cin + c] = (bf16_t)w[gid];
}

// Edge-conv factorization: BuT = (W1-W2) (O x C), BvT = W2 (O x C), bf16
__global__ __launch_bounds__(256) void k_prep_ec(
    const float* __restrict__ ecw, bf16_t* __restrict__ BuT,
    bf16_t* __restrict__ BvT, int C, int O)
{
    long long gid = (long long)blockIdx.x * blockDim.x + threadIdx.x;
    if (gid >= (long long)C * O) return;
    int c = (int)(gid % C);
    int o = (int)(gid / C);
    float w1 = ecw[(long long)o * (2 * C) + c];
    float w2 = ecw[(long long)o * (2 * C) + C + c];
    BuT[(long long)o * C + c] = (bf16_t)(w1 - w2);
    BvT[(long long)o * C + c] = (bf16_t)w2;
}

// ---------------------------------------------------------------------------
// Shared epilogue: y = act(BN(acc + bias)) + residual -> f32 and/or bf16
// ---------------------------------------------------------------------------
__device__ __forceinline__ void gemm_epilogue(
    v8f acc[2][2], int block_m, int block_n, int wm, int wn,
    int khalf, int frow, int M, int N,
    const float* bias, const float* bng, const float* bnb,
    const float* residual, int act,
    float* Cf, bf16_t* Cbf)
{
    #pragma unroll
    for (int tm = 0; tm < 2; ++tm)
        #pragma unroll
        for (int tn = 0; tn < 2; ++tn)
            #pragma unroll
            for (int r = 0; r < 8; ++r) {
                int mrow = block_m + wm + tm * 16 + khalf * 8 + r;
                int ncol = block_n + wn + tn * 16 + frow;
                if (mrow < M) {
                    float y = acc[tm][tn][r];
                    if (bias) y += bias[ncol];
                    if (bng)  y = y * (bng[ncol] * BN_INV) + bnb[ncol];
                    if (act == 1)      y = fmaxf(y, 0.0f);
                    else if (act == 2) y = gelu_exact(y);
                    if (residual) y += residual[(long long)mrow * N + ncol];
                    long long oi = (long long)mrow * N + ncol;
                    if (Cf)  Cf[oi]  = y;
                    if (Cbf) Cbf[oi] = (bf16_t)y;
                }
            }
}

// Fragment builders: contiguous-K LDS rows -> 2x ds_load_b128 per fragment
__device__ __forceinline__ v16bf load_frag(const bf16_t* rowp, int khalf)
{
    v16bf f;
    #pragma unroll
    for (int e = 0; e < 16; ++e)
        f[e] = rowp[khalf * 8 + (e < 8 ? e : 8 + e)];
    return f;
}

// ---------------------------------------------------------------------------
// 1x1 GEMM: C[M,N] = Abf[M,K] * BwT[N,K]^T. LDS tiles staged by the TDM with
// a double-buffered pipeline: wave 0 issues the NEXT tile pair, then waits
// s_wait_tensorcnt<=2 (previous pair retired, next pair in flight) so the
// DMA overlaps the WMMA work. 128 threads = 4 wave32, 64x64x32 tile.
// ---------------------------------------------------------------------------
__global__ __launch_bounds__(128) void k_gemm_1x1(
    const bf16_t* __restrict__ Abf, const bf16_t* __restrict__ BwT,
    float* __restrict__ Cf, bf16_t* __restrict__ Cbf,
    int M, int N, int K,
    const float* __restrict__ bias, const float* __restrict__ bng,
    const float* __restrict__ bnb, const float* __restrict__ residual,
    int act)
{
    __shared__ bf16_t As[2][TBM][TBK];   // LDS 0 .. 8191
    __shared__ bf16_t Bt[2][TBN][TBK];   // LDS 8192 .. 16383
    const unsigned SZ = (unsigned)(TBM * TBK * sizeof(bf16_t));  // 4096

    const int tid   = threadIdx.x;
    const int wave  = tid >> 5;
    const int lane  = tid & 31;
    const int frow  = lane & 15;
    const int khalf = lane >> 4;
    const int wm = (wave >> 1) * 32;
    const int wn = (wave & 1) * 32;
    const int block_m = blockIdx.x * TBM;
    const int block_n = blockIdx.y * TBN;

    const int ntiles = K / TBK;
    v8f acc[2][2] = {};

    // prologue: stage tile 0 into buffer 0
    if (wave == 0) {
        tdm_load_2d(0u,      Abf + (long long)block_m * K, K, M - block_m,
                    TBK, TBM, K);
        tdm_load_2d(2u * SZ, BwT + (long long)block_n * K, K, N - block_n,
                    TBK, TBN, K);
    }

    for (int t = 0; t < ntiles; ++t) {
        const int cur = t & 1;
        if (wave == 0) {
            if (t + 1 < ntiles) {
                const int nxt = (t + 1) & 1;
                int k1 = (t + 1) * TBK;
                tdm_load_2d((unsigned)nxt * SZ,
                            Abf + (long long)block_m * K + k1,
                            K - k1, M - block_m, TBK, TBM, K);
                tdm_load_2d(2u * SZ + (unsigned)nxt * SZ,
                            BwT + (long long)block_n * K + k1,
                            K - k1, N - block_n, TBK, TBN, K);
                __builtin_amdgcn_s_wait_tensorcnt(2);  // pair t retired
            } else {
                __builtin_amdgcn_s_wait_tensorcnt(0);  // drain tail
            }
        }
        // escape LDS arrays so the TDM writes aren't optimized away
        asm volatile("" ::"v"(&As[0][0][0]), "v"(&Bt[0][0][0]) : "memory");
        __syncthreads();

        v16bf af[2], bfr[2];
        #pragma unroll
        for (int tm = 0; tm < 2; ++tm)
            af[tm] = load_frag(&As[cur][wm + tm * 16 + frow][0], khalf);
        #pragma unroll
        for (int tn = 0; tn < 2; ++tn)
            bfr[tn] = load_frag(&Bt[cur][wn + tn * 16 + frow][0], khalf);
        #pragma unroll
        for (int tm = 0; tm < 2; ++tm)
            #pragma unroll
            for (int tn = 0; tn < 2; ++tn)
                acc[tm][tn] = __builtin_amdgcn_wmma_f32_16x16x32_bf16(
                    false, af[tm], false, bfr[tn], (short)0, acc[tm][tn],
                    false, false);
        __syncthreads();   // all waves done reading buf[cur] before reuse
    }
    gemm_epilogue(acc, block_m, block_n, wm, wn, khalf, frow, M, N,
                  bias, bng, bnb, residual, act, Cf, Cbf);
}

// ---------------------------------------------------------------------------
// 3x3 implicit-GEMM conv (division-free gather): A = bf16 NHWC activations.
// (ky,kx,c0) is uniform per 32-wide K tile (inC % 32 == 0); per-thread row
// decode hoisted out of the K loop.
// ---------------------------------------------------------------------------
__global__ __launch_bounds__(128) void k_gemm_conv3(
    const bf16_t* __restrict__ Abf, const bf16_t* __restrict__ BwT,
    float* __restrict__ Cf, bf16_t* __restrict__ Cbf,
    int M, int N, int K,
    int inH, int inW, int inC, int stride, int pad,
    int OW, int OHW,
    const float* __restrict__ bias, const float* __restrict__ bng,
    const float* __restrict__ bnb, int act)
{
    __shared__ bf16_t As[TBM][TBK];
    __shared__ bf16_t Bt[TBN][TBK];

    const int tid   = threadIdx.x;
    const int wave  = tid >> 5;
    const int lane  = tid & 31;
    const int frow  = lane & 15;
    const int khalf = lane >> 4;
    const int wm = (wave >> 1) * 32;
    const int wn = (wave & 1) * 32;
    const int block_m = blockIdx.x * TBM;
    const int block_n = blockIdx.y * TBN;
    const int kk = tid & 31;          // fixed K-lane for A staging

    // hoist per-thread row decode (invariant over K loop); 16 m-slots/thread
    int pb[16], piy[16], pix[16];
    #pragma unroll
    for (int it = 0; it < 16; ++it) {
        int row = block_m + it * 4 + wave;
        int b  = row / OHW; int rr = row - b * OHW;
        int oy = rr / OW;   int ox = rr - oy * OW;
        int iy0 = oy * stride - pad, ix0 = ox * stride - pad;
        piy[it] = iy0; pix[it] = ix0;
        pb[it]  = (b * inH + iy0) * inW + ix0;
    }

    v8f acc[2][2] = {};

    for (int k0 = 0; k0 < K; k0 += TBK) {
        // uniform (per K-tile) kernel-position decode: one scalar divide
        int kyx = k0 / inC;
        int c0  = k0 - kyx * inC;
        int ky  = kyx / 3, kx = kyx - ky * 3;
        // ---- stage A (implicit im2col, no per-element divides) ----
        #pragma unroll
        for (int it = 0; it < 16; ++it) {
            int iy = piy[it] + ky, ix = pix[it] + kx;
            bf16_t v = (bf16_t)0.0f;
            if ((unsigned)iy < (unsigned)inH && (unsigned)ix < (unsigned)inW)
                v = Abf[(long long)(pb[it] + ky * inW + kx) * inC + c0 + kk];
            As[it * 4 + wave][kk] = v;
        }
        // ---- stage B: 2 threads/row, 16B vector copies ----
        {
            int rown = tid >> 1, half = (tid & 1) * 16;
            const uint4* s = (const uint4*)(BwT +
                (long long)(block_n + rown) * K + k0 + half);
            *(uint4*)&Bt[rown][half]     = s[0];
            *(uint4*)&Bt[rown][half + 8] = s[1];
        }
        if (k0 + TBK < K)   // prefetch next B tile (global_prefetch_b8)
            __builtin_prefetch(BwT + (long long)(block_n + (tid >> 1)) * K +
                               k0 + TBK, 0, 3);
        __syncthreads();

        v16bf af[2], bfr[2];
        #pragma unroll
        for (int tm = 0; tm < 2; ++tm)
            af[tm] = load_frag(&As[wm + tm * 16 + frow][0], khalf);
        #pragma unroll
        for (int tn = 0; tn < 2; ++tn)
            bfr[tn] = load_frag(&Bt[wn + tn * 16 + frow][0], khalf);
        #pragma unroll
        for (int tm = 0; tm < 2; ++tm)
            #pragma unroll
            for (int tn = 0; tn < 2; ++tn)
                acc[tm][tn] = __builtin_amdgcn_wmma_f32_16x16x32_bf16(
                    false, af[tm], false, bfr[tn], (short)0, acc[tm][tn],
                    false, false);
        __syncthreads();
    }
    gemm_epilogue(acc, block_m, block_n, wm, wn, khalf, frow, M, N,
                  bias, bng, bnb, nullptr, act, Cf, Cbf);
}

// ---------------------------------------------------------------------------
// pos add: h[(b,n),c] += pos[c,n]; writes f32 (in place) + bf16 shadow
// ---------------------------------------------------------------------------
__global__ __launch_bounds__(256) void k_addpos(
    float* __restrict__ h, bf16_t* __restrict__ hbf,
    const float* __restrict__ pos, int Nsp, int C, long long total)
{
    long long gid = (long long)blockIdx.x * blockDim.x + threadIdx.x;
    if (gid >= total) return;
    int c = (int)(gid % C);
    long long row = gid / C;
    int n = (int)(row % Nsp);
    float y = h[gid] + pos[(long long)c * Nsp + n];
    h[gid] = y;
    hbf[gid] = (bf16_t)y;
}

// ---------------------------------------------------------------------------
// Row L2-normalize
// ---------------------------------------------------------------------------
__global__ __launch_bounds__(256) void k_rownorm(
    const float* __restrict__ f, float* __restrict__ fnn, int C)
{
    int row = blockIdx.x;
    __shared__ float red[256];
    float s = 0.0f;
    for (int i = threadIdx.x; i < C; i += 256) {
        float v = f[(long long)row * C + i];
        s += v * v;
    }
    red[threadIdx.x] = s;
    __syncthreads();
    for (int st = 128; st > 0; st >>= 1) {
        if (threadIdx.x < st) red[threadIdx.x] += red[threadIdx.x + st];
        __syncthreads();
    }
    float inv = 1.0f / (sqrtf(red[0]) + 1e-12f);
    for (int i = threadIdx.x; i < C; i += 256)
        fnn[(long long)row * C + i] = f[(long long)row * C + i] * inv;
}

// ---------------------------------------------------------------------------
// Top-9 cosine neighbors (absolute row indices)
// ---------------------------------------------------------------------------
__global__ __launch_bounds__(256) void k_topk(
    const float* __restrict__ fnn, int* __restrict__ idx, int Nsp, int C)
{
    int row  = blockIdx.x;
    int base = (row / Nsp) * Nsp;
    __shared__ float srow[768];
    for (int i = threadIdx.x; i < C; i += 256)
        srow[i] = fnn[(long long)row * C + i];
    __syncthreads();

    float bv[TOPK]; int bi[TOPK];
    #pragma unroll
    for (int t = 0; t < TOPK; ++t) { bv[t] = -1e30f; bi[t] = base; }

    for (int m = threadIdx.x; m < Nsp; m += 256) {
        const float* other = fnn + (long long)(base + m) * C;
        float s = 0.0f;
        for (int c = 0; c < C; ++c) s += srow[c] * other[c];
        if (s > bv[TOPK - 1]) {
            int p = TOPK - 1;
            while (p > 0 && bv[p - 1] < s) {
                bv[p] = bv[p - 1]; bi[p] = bi[p - 1]; --p;
            }
            bv[p] = s; bi[p] = base + m;
        }
    }
    __shared__ float mv[256 * TOPK];
    __shared__ int   mi[256 * TOPK];
    #pragma unroll
    for (int t = 0; t < TOPK; ++t) {
        mv[threadIdx.x * TOPK + t] = bv[t];
        mi[threadIdx.x * TOPK + t] = bi[t];
    }
    __syncthreads();
    if (threadIdx.x == 0) {
        for (int t = 0; t < TOPK; ++t) {
            float best = -1e30f; int bidx = base, bpos = 0;
            for (int j = 0; j < 256 * TOPK; ++j)
                if (mv[j] > best) { best = mv[j]; bidx = mi[j]; bpos = j; }
            mv[bpos] = -1e31f;
            idx[row * TOPK + t] = bidx;
        }
    }
}

// ---------------------------------------------------------------------------
// Edge gather-max -> bf16 (A matrix of gf2):
// emax = gelu((u + max_k v[idx_k]) * s + t); exact: s>0, gelu monotone
// ---------------------------------------------------------------------------
__global__ __launch_bounds__(256) void k_gathermax(
    const float* __restrict__ u, const float* __restrict__ v,
    const int* __restrict__ idx, const float* __restrict__ ecg,
    const float* __restrict__ ecbb, bf16_t* __restrict__ emax,
    long long total, int Cout)
{
    long long gid = (long long)blockIdx.x * blockDim.x + threadIdx.x;
    if (gid >= total) return;
    int o = (int)(gid % Cout);
    long long row = gid / Cout;
    float m = -1e30f;
    #pragma unroll
    for (int k = 0; k < TOPK; ++k) {
        int j = idx[row * TOPK + k];
        m = fmaxf(m, v[(long long)j * Cout + o]);
    }
    float e = u[gid] + m;
    e = e * (ecg[o] * BN_INV) + ecbb[o];
    emax[gid] = (bf16_t)gelu_exact(e);
}

// ---------------------------------------------------------------------------
// Global max pool + head
// ---------------------------------------------------------------------------
__global__ __launch_bounds__(256) void k_maxpool(
    const float* __restrict__ g, float* __restrict__ p,
    int Bn, int Nsp, int C)
{
    int gid = blockIdx.x * blockDim.x + threadIdx.x;
    if (gid >= Bn * C) return;
    int c = gid % C;
    int b = gid / C;
    float m = -1e30f;
    for (int n = 0; n < Nsp; ++n)
        m = fmaxf(m, g[((long long)b * Nsp + n) * C + c]);
    p[gid] = m;
}

__global__ __launch_bounds__(256) void k_head(
    const float* __restrict__ in, const float* __restrict__ w,
    const float* __restrict__ bias, const float* __restrict__ g,
    const float* __restrict__ bb, float* __restrict__ out,
    int Bn, int Cin, int Cout, int act)
{
    int gid = blockIdx.x * blockDim.x + threadIdx.x;
    if (gid >= Bn * Cout) return;
    int o = gid % Cout;
    int b = gid / Cout;
    float acc = bias[o];
    for (int c = 0; c < Cin; ++c)
        acc += in[b * Cin + c] * w[(long long)o * Cin + c];
    if (g) acc = acc * (g[o] * BN_INV) + bb[o];
    if (act == 1) acc = fmaxf(acc, 0.0f);
    out[gid] = acc;
}

// ---------------------------------------------------------------------------
// Host orchestration
// ---------------------------------------------------------------------------
static inline void launch_1x1(hipStream_t s, const bf16_t* A, const bf16_t* BwT,
                              float* Cf, bf16_t* Cbf, int M, int N, int K,
                              const float* bias, const float* bng,
                              const float* bnb, const float* res, int act)
{
    dim3 grid(M / TBM, N / TBN);
    k_gemm_1x1<<<grid, 128, 0, s>>>(A, BwT, Cf, Cbf, M, N, K,
                                    bias, bng, bnb, res, act);
}

extern "C" void kernel_launch(void* const* d_in, const int* in_sizes, int n_in,
                              void* d_out, int out_size, void* d_ws,
                              size_t ws_size, hipStream_t stream)
{
    (void)in_sizes; (void)n_in; (void)out_size; (void)ws_size;
    const float* x    = (const float*)d_in[0];
    const float* s1w  = (const float*)d_in[1];
    const float* s1b  = (const float*)d_in[2];
    const float* bn1g = (const float*)d_in[3];
    const float* bn1b = (const float*)d_in[4];
    const float* s2w  = (const float*)d_in[5];
    const float* s2b  = (const float*)d_in[6];
    const float* bn2g = (const float*)d_in[7];
    const float* bn2b = (const float*)d_in[8];
    const float* s3w  = (const float*)d_in[9];
    const float* s3b  = (const float*)d_in[10];
    const float* bn3g = (const float*)d_in[11];
    const float* bn3b = (const float*)d_in[12];
    const float* pos  = (const float*)d_in[13];
    const float* gf1w = (const float*)d_in[14];
    const float* gf1b = (const float*)d_in[15];
    const float* gf1g = (const float*)d_in[16];
    const float* gf1bb= (const float*)d_in[17];
    const float* ecw  = (const float*)d_in[18];
    const float* ecb  = (const float*)d_in[19];
    const float* ecg  = (const float*)d_in[20];
    const float* ecbb = (const float*)d_in[21];
    const float* gf2w = (const float*)d_in[22];
    const float* gf2b = (const float*)d_in[23];
    const float* gf2g = (const float*)d_in[24];
    const float* gf2bb= (const float*)d_in[25];
    const float* ff1w = (const float*)d_in[26];
    const float* ff1b = (const float*)d_in[27];
    const float* ff1g = (const float*)d_in[28];
    const float* ff1bb= (const float*)d_in[29];
    const float* ff2w = (const float*)d_in[30];
    const float* ff2b = (const float*)d_in[31];
    const float* ff2g = (const float*)d_in[32];
    const float* ff2bb= (const float*)d_in[33];
    const float* c1w  = (const float*)d_in[34];
    const float* c1b  = (const float*)d_in[35];
    const float* cbg  = (const float*)d_in[36];
    const float* cbb  = (const float*)d_in[37];
    const float* c2w  = (const float*)d_in[38];
    const float* c2b  = (const float*)d_in[39];
    float* out = (float*)d_out;

    const int Bn = 4, Nsp = 56 * 56;       // 3136
    const int Mtot = Bn * Nsp;             // 12544

    char* ws = (char*)d_ws;
    size_t off = 0;
    auto alloc = [&](size_t bytes) -> void* {
        void* p = ws + off;
        off = (off + bytes + 255) & ~(size_t)255;
        return p;
    };
    bf16_t* h1_bf   = (bf16_t*)alloc((size_t)Bn * 112 * 112 * 384 * 2); // 38.5MB
    bf16_t* h2_bf   = (bf16_t*)alloc((size_t)Mtot * 768 * 2);
    float*  hpos    = (float*) alloc((size_t)Mtot * 768 * 4);
    bf16_t* hpos_bf = (bf16_t*)alloc((size_t)Mtot * 768 * 2);
    float*  f       = (float*) alloc((size_t)Mtot * 768 * 4);
    bf16_t* f_bf    = (bf16_t*)alloc((size_t)Mtot * 768 * 2);
    float*  fnn     = (float*) alloc((size_t)Mtot * 768 * 4);
    float*  u       = (float*) alloc((size_t)Mtot * 1536 * 4);
    float*  v       = (float*) alloc((size_t)Mtot * 1536 * 4);
    bf16_t* BwT     = (bf16_t*)alloc((size_t)768 * 6912 * 2);
    bf16_t* BuT     = (bf16_t*)alloc((size_t)1536 * 768 * 2);
    bf16_t* BvT     = (bf16_t*)alloc((size_t)1536 * 768 * 2);
    int*    idx     = (int*)   alloc((size_t)Mtot * TOPK * 4);
    float*  p       = (float*) alloc((size_t)Bn * 768 * 4);
    float*  q       = (float*) alloc((size_t)Bn * 1024 * 4);
    // dead-buffer reuse
    bf16_t* emax_bf = h1_bf;              // h1 dead after s2; 38.5MB each
    float*  g       = u;                  // u dead after gathermax
    bf16_t* g_bf    = (bf16_t*)v;         // v dead after gathermax
    bf16_t* ff1o_bf = h2_bf;              // h2 dead after s3
    float*  gf      = fnn;                // fnn dead after topk

    const int T = 256;
    auto blocks = [](long long n) { return (unsigned)((n + 255) / 256); };

    // ---- stem 1 (scalar, K=27) -> h1 NHWC bf16 ----
    k_stem1<<<blocks((long long)Bn * 112 * 112 * 384), T, 0, stream>>>(
        x, s1w, s1b, bn1g, bn1b, h1_bf, Bn, 3, 224, 224, 384, 112, 112);

    // ---- stem 2: 384->768 3x3 s2 (WMMA) -> h2 bf16 ----
    k_prep_w<<<blocks((long long)768 * 384 * 9), T, 0, stream>>>(s2w, BwT, 768, 384, 3);
    k_gemm_conv3<<<dim3(Mtot / TBM, 768 / TBN), 128, 0, stream>>>(
        h1_bf, BwT, nullptr, h2_bf, Mtot, 768, 384 * 9,
        112, 112, 384, 2, 1, 56, Nsp, s2b, bn2g, bn2b, 1);

    // ---- stem 3: 768->768 3x3 s1 (WMMA) -> hpos f32 ----
    k_prep_w<<<blocks((long long)768 * 768 * 9), T, 0, stream>>>(s3w, BwT, 768, 768, 3);
    k_gemm_conv3<<<dim3(Mtot / TBM, 768 / TBN), 128, 0, stream>>>(
        h2_bf, BwT, hpos, nullptr, Mtot, 768, 768 * 9,
        56, 56, 768, 1, 1, 56, Nsp, s3b, bn3g, bn3b, 0);

    // ---- + pos (f32 in place + bf16 shadow) ----
    k_addpos<<<blocks((long long)Mtot * 768), T, 0, stream>>>(
        hpos, hpos_bf, pos, Nsp, 768, (long long)Mtot * 768);

    // ---- grapher fc1 (TDM + WMMA) -> f f32 + bf16 ----
    k_prep_w<<<blocks((long long)768 * 768), T, 0, stream>>>(gf1w, BwT, 768, 768, 1);
    launch_1x1(stream, hpos_bf, BwT, f, f_bf, Mtot, 768, 768,
               gf1b, gf1g, gf1bb, nullptr, 0);

    // ---- normalize + top-9 KNN ----
    k_rownorm<<<Mtot, T, 0, stream>>>(f, fnn, 768);
    k_topk<<<Mtot, T, 0, stream>>>(fnn, idx, Nsp, 768);

    // ---- edge factors u = f(W1-W2)+ecb, v = f W2 (TDM + WMMA) ----
    k_prep_ec<<<blocks((long long)768 * 1536), T, 0, stream>>>(ecw, BuT, BvT, 768, 1536);
    launch_1x1(stream, f_bf, BuT, u, nullptr, Mtot, 1536, 768,
               ecb, nullptr, nullptr, nullptr, 0);
    launch_1x1(stream, f_bf, BvT, v, nullptr, Mtot, 1536, 768,
               nullptr, nullptr, nullptr, nullptr, 0);

    // ---- gather-max + affine + gelu -> emax bf16 ----
    k_gathermax<<<blocks((long long)Mtot * 1536), T, 0, stream>>>(
        u, v, idx, ecg, ecbb, emax_bf, (long long)Mtot * 1536, 1536);

    // ---- grapher fc2: 1536->768 + BN + residual (TDM + WMMA) ----
    k_prep_w<<<blocks((long long)768 * 1536), T, 0, stream>>>(gf2w, BwT, 768, 1536, 1);
    launch_1x1(stream, emax_bf, BwT, g, g_bf, Mtot, 768, 1536,
               gf2b, gf2g, gf2bb, hpos, 0);

    // ---- FFN (TDM + WMMA) ----
    k_prep_w<<<blocks((long long)768 * 768), T, 0, stream>>>(ff1w, BwT, 768, 768, 1);
    launch_1x1(stream, g_bf, BwT, nullptr, ff1o_bf, Mtot, 768, 768,
               ff1b, ff1g, ff1bb, nullptr, 2);
    k_prep_w<<<blocks((long long)768 * 768), T, 0, stream>>>(ff2w, BwT, 768, 768, 1);
    launch_1x1(stream, ff1o_bf, BwT, gf, nullptr, Mtot, 768, 768,
               ff2b, ff2g, ff2bb, g, 0);

    // ---- pool + head ----
    k_maxpool<<<blocks((long long)Bn * 768), T, 0, stream>>>(gf, p, Bn, Nsp, 768);
    k_head<<<blocks((long long)Bn * 1024), T, 0, stream>>>(
        p, c1w, c1b, cbg, cbb, q, Bn, 768, 1024, 1);
    k_head<<<blocks((long long)Bn * 1000), T, 0, stream>>>(
        q, c2w, c2b, nullptr, nullptr, out, Bn, 1024, 1000, 0);
}
